// RecurrentSig_89103391522933
// MI455X (gfx1250) — matrix operand order repaired and checked by reading, not compiled, not run on hardware
//
#include <hip/hip_runtime.h>

// ---------------------------------------------------------------------------
// Problem constants (from the reference)
// ---------------------------------------------------------------------------
#define BATCH   4096
#define NUNITS  1024
#define KX      512                // x / W_w inner dim
#define KSIG    6144               // prev_sigs width (1024 units * 6)
#define KSTATE  1024               // prev_states width
#define KTOT    7680               // 512 + 6144 + 1024
#define SAVED   7168               // U_w inner dim (6144 + 1024)
#define OUTROW  7168               // out row: 6144 sig + 1024 raw
#define SIGOFF  6144

// GEMM tiling
#define BM      128
#define BN      128
#define KB      32                 // K-slab per LDS stage (== WMMA K)
#define LSTR    40                 // LDS row stride in bf16 elems (80B, padded)

typedef __attribute__((ext_vector_type(8)))  float  v8f;
typedef __attribute__((ext_vector_type(16))) __bf16 v16bf;
typedef __attribute__((ext_vector_type(8)))  __bf16 v8bf;
typedef __attribute__((ext_vector_type(4)))  __bf16 v4bf;

union Frag16 { v16bf v; v8bf h[2]; };

// f32 -> bf16 hi/lo split (hi = rne(x), lo = rne(x - hi))
__device__ __forceinline__ void cvt_split4(float4 v, v4bf& h, v4bf& l) {
  __bf16 h0 = (__bf16)v.x, h1 = (__bf16)v.y, h2 = (__bf16)v.z, h3 = (__bf16)v.w;
  h = (v4bf){h0, h1, h2, h3};
  l = (v4bf){(__bf16)(v.x - (float)h0), (__bf16)(v.y - (float)h1),
             (__bf16)(v.z - (float)h2), (__bf16)(v.w - (float)h3)};
}

// ---------------------------------------------------------------------------
// One K-segment with compile-time strides. aRow/bRow are this thread's
// staging pointers (row srow, col scol already applied). The 4 staging rows
// are at constant byte offsets rr*32*STRIDE*4 (< 8MB, folds into the load's
// 24-bit immediate). Software pipeline: slab kt+1 is loaded into registers
// while slab kt computes out of LDS.
// ---------------------------------------------------------------------------
template <int ASTR, int BSTR>
__device__ __forceinline__ void run_segment(
    const float* __restrict__ aRow, const float* __restrict__ bRow,
    const int nSlabs,
    __bf16* Ahi, __bf16* Alo, __bf16* Bhi, __bf16* Blo,
    const int srow, const int scol,
    const int l16, const int lhalf, const int waveM, const int waveN,
    v8f (&acc)[2][4]) {
  float4 avr[4], bvr[4];
#pragma unroll
  for (int rr = 0; rr < 4; ++rr) {
    avr[rr] = *(const float4*)(aRow + rr * 32 * ASTR);
    bvr[rr] = *(const float4*)(bRow + rr * 32 * BSTR);
  }

  for (int kt = 0; kt < nSlabs; ++kt) {
    __syncthreads();   // previous slab fully consumed before overwrite

    // Convert the prefetched registers and commit to LDS
#pragma unroll
    for (int rr = 0; rr < 4; ++rr) {
      const int r = srow + rr * 32;
      v4bf ah, al, bh, bl;
      cvt_split4(avr[rr], ah, al);
      cvt_split4(bvr[rr], bh, bl);
      *(v4bf*)&Ahi[r * LSTR + scol] = ah;
      *(v4bf*)&Alo[r * LSTR + scol] = al;
      *(v4bf*)&Bhi[r * LSTR + scol] = bh;
      *(v4bf*)&Blo[r * LSTR + scol] = bl;
    }

    __syncthreads();

    // Issue global loads for slab kt+1; they retire while we compute.
    aRow += KB;
    bRow += KB;
    if (kt + 1 < nSlabs) {
#pragma unroll
      for (int rr = 0; rr < 4; ++rr) {
        avr[rr] = *(const float4*)(aRow + rr * 32 * ASTR);
        bvr[rr] = *(const float4*)(bRow + rr * 32 * BSTR);
      }
    }

    // Load fragments (bf16 16-bit A layout: lane half selects K sub-ranges)
    Frag16 fah[2], fal[2], fbh[4], fbl[4];
#pragma unroll
    for (int i = 0; i < 2; ++i) {
      const int row = waveM + i * 16 + l16;
      const __bf16* bh_ = &Ahi[row * LSTR];
      const __bf16* bl_ = &Alo[row * LSTR];
      fah[i].h[0] = *(const v8bf*)(bh_ + lhalf * 8);
      fah[i].h[1] = *(const v8bf*)(bh_ + 16 + lhalf * 8);
      fal[i].h[0] = *(const v8bf*)(bl_ + lhalf * 8);
      fal[i].h[1] = *(const v8bf*)(bl_ + 16 + lhalf * 8);
    }
#pragma unroll
    for (int j = 0; j < 4; ++j) {
      const int row = waveN + j * 16 + l16;
      const __bf16* bh_ = &Bhi[row * LSTR + lhalf * 8];
      const __bf16* bl_ = &Blo[row * LSTR + lhalf * 8];
      fbh[j].h[0] = *(const v8bf*)(bh_);
      fbh[j].h[1] = *(const v8bf*)(bh_ + 16);
      fbl[j].h[0] = *(const v8bf*)(bl_);
      fbl[j].h[1] = *(const v8bf*)(bl_ + 16);
    }

    // bf16x3: term-major order so consecutive WMMAs hit different
    // accumulators (8-instruction gap before any D-register reuse).
#pragma unroll
    for (int t = 0; t < 3; ++t)
#pragma unroll
      for (int i = 0; i < 2; ++i)
#pragma unroll
        for (int j = 0; j < 4; ++j) {
          const v16bf& A = (t == 2) ? fal[i].v : fah[i].v;
          const v16bf& B = (t == 1) ? fbl[j].v : fbh[j].v;
          acc[i][j] = __builtin_amdgcn_wmma_f32_16x16x32_bf16(
              false, A, false, B, (short)0, acc[i][j], false, false);
        }
  }
}

// ---------------------------------------------------------------------------
// Fused GEMM: raw = x@W_w^T + [sigs|states]@U_w^T + (W_b + U_b)
// bf16x3 split-precision WMMA, f32 accumulate. Writes raw into out[:, 6144:].
// ---------------------------------------------------------------------------
__global__ __launch_bounds__(256, 1)
void gemm_raw_kernel(const float* __restrict__ x,
                     const float* __restrict__ prev_sigs,
                     const float* __restrict__ prev_states,
                     const float* __restrict__ Ww,
                     const float* __restrict__ Wb,
                     const float* __restrict__ Uw,
                     const float* __restrict__ Ub,
                     float* __restrict__ out) {
  __shared__ __align__(16) __bf16 Ahi[BM * LSTR];
  __shared__ __align__(16) __bf16 Alo[BM * LSTR];
  __shared__ __align__(16) __bf16 Bhi[BN * LSTR];
  __shared__ __align__(16) __bf16 Blo[BN * LSTR];

  const int tid   = threadIdx.x;
  const int lane  = tid & 31;
  const int wave  = tid >> 5;          // 0..7
  const int l16   = lane & 15;
  const int lhalf = lane >> 4;         // 0/1

  const int M0 = blockIdx.y * BM;      // batch tile
  const int N0 = blockIdx.x * BN;      // unit tile
  const int waveM = (wave >> 1) * 32;  // 4 wave-rows of 32
  const int waveN = (wave & 1) * 64;   // 2 wave-cols of 64

  // Staging assignment: thread -> 4 rows x one float4 column chunk
  const int srow = tid >> 3;           // 0..31
  const int scol = (tid & 7) * 4;      // 0..28

  // Accumulators: 2 (M) x 4 (N) subtiles of 16x16, f32
  v8f acc[2][4];
#pragma unroll
  for (int j = 0; j < 4; ++j) {
    const int n = N0 + waveN + j * 16 + l16;
    const float bias = Wb[n] + Ub[n];
#pragma unroll
    for (int i = 0; i < 2; ++i)
#pragma unroll
      for (int e = 0; e < 8; ++e) acc[i][j][e] = bias;
  }

  // Three K-segments with compile-time strides:
  //   slabs [  0, 16): A = x           (512), B = W_w          (512)
  //   slabs [ 16,208): A = prev_sigs  (6144), B = U_w[:, 0:]  (7168)
  //   slabs [208,240): A = prev_states(1024), B = U_w[:,6144:](7168)
  run_segment<KX, KX>(
      x  + (size_t)(M0 + srow) * KX + scol,
      Ww + (size_t)(N0 + srow) * KX + scol,
      KX / KB,
      Ahi, Alo, Bhi, Blo, srow, scol, l16, lhalf, waveM, waveN, acc);

  run_segment<KSIG, SAVED>(
      prev_sigs + (size_t)(M0 + srow) * KSIG + scol,
      Uw        + (size_t)(N0 + srow) * SAVED + scol,
      KSIG / KB,
      Ahi, Alo, Bhi, Blo, srow, scol, l16, lhalf, waveM, waveN, acc);

  run_segment<KSTATE, SAVED>(
      prev_states + (size_t)(M0 + srow) * KSTATE + scol,
      Uw          + (size_t)(N0 + srow) * SAVED + KSIG + scol,
      KSTATE / KB,
      Ahi, Alo, Bhi, Blo, srow, scol, l16, lhalf, waveM, waveN, acc);

  // Store raw into out[:, SIGOFF:]  (C layout: vgpr v -> M = v + 8*lhalf)
#pragma unroll
  for (int i = 0; i < 2; ++i)
#pragma unroll
    for (int j = 0; j < 4; ++j) {
      const int n = N0 + waveN + j * 16 + l16;
#pragma unroll
      for (int v = 0; v < 8; ++v) {
        const int m = M0 + waveM + i * 16 + lhalf * 8 + v;
        out[(size_t)m * OUTROW + SIGOFF + n] = acc[i][j][v];
      }
    }
}

// ---------------------------------------------------------------------------
// Elementwise sigjoin (Chen identity, level 2) over 4096x1024 units.
// Reads raw from out[:,6144:], writes sigs into out[:,:6144].
// ---------------------------------------------------------------------------
__global__ __launch_bounds__(256)
void sigjoin_kernel(const float* __restrict__ prev_sigs,
                    const float* __restrict__ prev_states,
                    const float* __restrict__ log_tl,
                    float* __restrict__ out) {
  const int idx = blockIdx.x * 256 + threadIdx.x;     // b*1024 + u
  const int b = idx >> 10;
  const int u = idx & 1023;

  const float tl  = __expf(log_tl[0]);
  const float raw = out[(size_t)b * OUTROW + SIGOFF + u];
  const float ps  = prev_states[idx];
  const float d   = raw - ps;

  const float2* S = (const float2*)(prev_sigs + (size_t)b * KSIG + u * 6);
  const float2 s01 = S[0];   // level-1: S0, S1
  const float2 s23 = S[1];   // level-2: S2, S3
  const float2 s45 = S[2];   // level-2: S4, S5

  const float o0 = s01.x + d;
  const float o1 = s01.y + tl;
  const float o2 = s23.x + 0.5f * d * d   + s01.x * d;
  const float o3 = s23.y + 0.5f * d * tl  + s01.x * tl;
  const float o4 = s45.x + 0.5f * tl * d  + s01.y * d;
  const float o5 = s45.y + 0.5f * tl * tl + s01.y * tl;

  float2* O = (float2*)(out + (size_t)b * OUTROW + u * 6);
  O[0] = make_float2(o0, o1);
  O[1] = make_float2(o2, o3);
  O[2] = make_float2(o4, o5);
}

// ---------------------------------------------------------------------------
extern "C" void kernel_launch(void* const* d_in, const int* in_sizes, int n_in,
                              void* d_out, int out_size, void* d_ws, size_t ws_size,
                              hipStream_t stream) {
  const float* x           = (const float*)d_in[0];
  const float* prev_sigs   = (const float*)d_in[1];
  const float* prev_states = (const float*)d_in[2];
  const float* Ww          = (const float*)d_in[3];
  const float* Wb          = (const float*)d_in[4];
  const float* Uw          = (const float*)d_in[5];
  const float* Ub          = (const float*)d_in[6];
  const float* log_tl      = (const float*)d_in[7];
  float* out = (float*)d_out;

  dim3 grid(NUNITS / BN, BATCH / BM);   // (8, 32)
  gemm_raw_kernel<<<grid, 256, 0, stream>>>(x, prev_sigs, prev_states,
                                            Ww, Wb, Uw, Ub, out);

  const int total = BATCH * NUNITS;
  sigjoin_kernel<<<total / 256, 256, 0, stream>>>(prev_sigs, prev_states,
                                                  log_tl, out);
}
